// HierarchicalProcessor_63934883168289
// MI455X (gfx1250) — matrix operand order repaired and checked by reading, not compiled, not run on hardware
//
#include <hip/hip_runtime.h>

#define D 128

typedef __attribute__((ext_vector_type(16))) __bf16          v16bf;
typedef __attribute__((ext_vector_type(16))) unsigned short  v16us;
typedef __attribute__((ext_vector_type(8)))  float           v8f;

__device__ __forceinline__ unsigned short f2bf(float f) {
  unsigned u = __builtin_bit_cast(unsigned, f);
  u += 0x7FFFu + ((u >> 16) & 1u);           // round-to-nearest-even
  return (unsigned short)(u >> 16);
}

__device__ __forceinline__ float siluf(float x) { return x / (1.0f + __expf(-x)); }

// ---------------------------------------------------------------------------
// Core 16xK * Kx128 tile GEMM using V_WMMA_F32_16X16X32_BF16.
// A tile lives in LDS (bf16, row stride `krow` ushorts). Weights are
// pre-swizzled so each lane's B fragment is 16 contiguous ushorts.
// acc layout (C/D): VGPR r, lane l -> row = r + 8*(l>>4), col = nt*16 + (l&15)
// ---------------------------------------------------------------------------
__device__ __forceinline__ void wmma_gemm(const unsigned short* sAb, int krow, int K,
                                          const v16us* __restrict__ wz,
                                          const float* __restrict__ bias,
                                          v8f acc[8], int lane)
{
  const int half = lane >> 4;
  const int l15  = lane & 15;
  #pragma unroll
  for (int nt = 0; nt < 8; ++nt) {
    float bv = bias[nt * 16 + l15];
    #pragma unroll
    for (int r = 0; r < 8; ++r) acc[nt][r] = bv;
  }
  const int nch = K >> 5;
  for (int kc = 0; kc < nch; ++kc) {
    // A fragment: contiguous k order, same as B swizzle: K = kc*32 + half*16 + i
    const unsigned short* ar = sAb + l15 * krow + (kc << 5) + half * 16;
    v16us ua;
    #pragma unroll
    for (int i = 0; i < 8; ++i)  ua[i] = ar[i];
    #pragma unroll
    for (int i = 8; i < 16; ++i) ua[i] = ar[i];
    v16bf a = __builtin_bit_cast(v16bf, ua);
    #pragma unroll
    for (int nt = 0; nt < 8; ++nt) {
      v16us ub = wz[(kc * 8 + nt) * 32 + lane];
      v16bf b  = __builtin_bit_cast(v16bf, ub);
      acc[nt] = __builtin_amdgcn_wmma_f32_16x16x32_bf16(
          false, a, false, b, (short)0, acc[nt], false, false);
    }
  }
}

// SiLU + write back to LDS as bf16 in [16][128] layout for the next layer.
__device__ __forceinline__ void act_store(const v8f acc[8], unsigned short* sOut, int lane)
{
  const int half = lane >> 4, l15 = lane & 15;
  #pragma unroll
  for (int nt = 0; nt < 8; ++nt) {
    int col = nt * 16 + l15;
    #pragma unroll
    for (int r = 0; r < 8; ++r) {
      float v = siluf(acc[nt][r]);
      sOut[(r + (half << 3)) * D + col] = f2bf(v);
    }
  }
}

// Store final-layer fp32 into LDS for LayerNorm.
__device__ __forceinline__ void f32_store(const v8f acc[8], float* sF, int lane)
{
  const int half = lane >> 4, l15 = lane & 15;
  #pragma unroll
  for (int nt = 0; nt < 8; ++nt) {
    int col = nt * 16 + l15;
    #pragma unroll
    for (int r = 0; r < 8; ++r) sF[(r + (half << 3)) * D + col] = acc[nt][r];
  }
}

// ---------------------------------------------------------------------------
// Weight swizzle: fp32 W[K][128] -> bf16 fragments.
// out[((kc*8+nt)*32+lane)*16+i] = bf16(W[kc*32 + 16*(lane>>4) + i][nt*16 + (lane&15)])
// ---------------------------------------------------------------------------
__global__ void swizzle_k(const float* __restrict__ W, unsigned short* __restrict__ out, int K)
{
  int t = blockIdx.x * blockDim.x + threadIdx.x;
  if (t >= K * D) return;
  int i    = t & 15;
  int lane = (t >> 4) & 31;
  int nt   = (t >> 9) & 7;
  int kc   = t >> 12;
  int col  = nt * 16 + (lane & 15);
  int row  = (kc << 5) + ((lane >> 4) << 4) + i;
  out[t] = f2bf(W[(size_t)row * D + col]);
}

__global__ void zero_k(float4* __restrict__ p, long n)
{
  long i = (long)blockIdx.x * blockDim.x + threadIdx.x;
  if (i < n) p[i] = make_float4(0.f, 0.f, 0.f, 0.f);
}

__global__ void copy_k(float4* __restrict__ dst, const float4* __restrict__ src, long n)
{
  long i = (long)blockIdx.x * blockDim.x + threadIdx.x;
  if (i < n) dst[i] = src[i];
}

// ---------------------------------------------------------------------------
// Edge MLP: per 16-edge tile: gather cat(edge_attr, send[src], rec[dst]) ->
// 3-layer MLP (WMMA bf16) + LayerNorm -> atomic scatter-add into aggr[dst].
// ---------------------------------------------------------------------------
__global__ __launch_bounds__(32) void edge_mlp_k(
    const int* __restrict__ ei, const float* __restrict__ ea,
    const float* __restrict__ fsend, const float* __restrict__ frec,
    const v16us* __restrict__ w1, const v16us* __restrict__ w2, const v16us* __restrict__ w3,
    const float* __restrict__ b1, const float* __restrict__ b2, const float* __restrict__ b3,
    const float* __restrict__ lng, const float* __restrict__ lnb,
    float* __restrict__ aggr, int E)
{
  __shared__ __align__(16) unsigned short sA[16 * 384];   // 12 KB, reused per layer
  __shared__ int sidx[32];                                // [0..15]=src, [16..31]=dst
  const int lane = threadIdx.x;
  const int base = blockIdx.x << 4;

  if (lane < 16) {
    sidx[lane]      = ei[base + lane];
    sidx[16 + lane] = ei[E + base + lane];
  }
  __syncthreads();

  // Gather [16][384] fp32 -> bf16 LDS (coalesced float4 loads)
  #pragma unroll 4
  for (int it = 0; it < 48; ++it) {
    int idx  = lane + (it << 5);       // 0..1535
    int row  = idx / 96;
    int c    = idx - row * 96;
    int seg  = c >> 5, col4 = c & 31;
    const float* p;
    if (seg == 0)      p = ea    + (size_t)(base + row) * D;
    else if (seg == 1) p = fsend + (size_t)sidx[row] * D;
    else               p = frec  + (size_t)sidx[16 + row] * D;
    float4 v = ((const float4*)p)[col4];
    unsigned short* o = sA + row * 384 + (seg << 7) + (col4 << 2);
    o[0] = f2bf(v.x); o[1] = f2bf(v.y); o[2] = f2bf(v.z); o[3] = f2bf(v.w);
  }
  __syncthreads();

  v8f acc[8];
  wmma_gemm(sA, 384, 384, w1, b1, acc, lane);
  __syncthreads(); act_store(acc, sA, lane); __syncthreads();
  wmma_gemm(sA, 128, 128, w2, b2, acc, lane);
  __syncthreads(); act_store(acc, sA, lane); __syncthreads();
  wmma_gemm(sA, 128, 128, w3, b3, acc, lane);
  __syncthreads();

  float* sF = (float*)sA;
  f32_store(acc, sF, lane);
  __syncthreads();

  // LayerNorm over D per row, then scatter-add message to aggr[dst]
  const int half = lane >> 4, rowm = lane & 15;
  const float* rp = sF + rowm * D + (half << 6);
  float s = 0.f, s2 = 0.f;
  #pragma unroll 8
  for (int c = 0; c < 64; ++c) { float v = rp[c]; s += v; s2 += v * v; }
  s  += __shfl_xor(s,  16, 32);
  s2 += __shfl_xor(s2, 16, 32);
  float mu   = s * (1.f / 128.f);
  float rstd = rsqrtf(s2 * (1.f / 128.f) - mu * mu + 1e-5f);

  float* ap = aggr + (size_t)sidx[16 + rowm] * D;
  for (int c = 0; c < 64; ++c) {
    int col = (half << 6) + c;
    float v = (rp[c] - mu) * rstd * lng[col] + lnb[col];
    atomicAdd(ap + col, v);
  }
}

// ---------------------------------------------------------------------------
// Node (aggr) MLP: in = cat(aggr[n], rec[n]) [16][256]; out = alpha*rec + diff
// ---------------------------------------------------------------------------
__global__ __launch_bounds__(32) void node_mlp_k(
    const float* __restrict__ aggb, const float* frec,
    const v16us* __restrict__ w1, const v16us* __restrict__ w2, const v16us* __restrict__ w3,
    const float* __restrict__ b1, const float* __restrict__ b2, const float* __restrict__ b3,
    const float* __restrict__ lng, const float* __restrict__ lnb,
    float* fout, float alpha)
{
  __shared__ __align__(16) unsigned short sA[16 * 384];
  const int lane = threadIdx.x;
  const int base = blockIdx.x << 4;

  #pragma unroll 4
  for (int it = 0; it < 32; ++it) {
    int idx = lane + (it << 5);        // 0..1023
    int row = idx >> 6;
    int c   = idx & 63;
    int seg = c >> 5, col4 = c & 31;
    const float* p = (seg ? frec : aggb) + (size_t)(base + row) * D;
    float4 v = ((const float4*)p)[col4];
    unsigned short* o = sA + row * 256 + (seg << 7) + (col4 << 2);
    o[0] = f2bf(v.x); o[1] = f2bf(v.y); o[2] = f2bf(v.z); o[3] = f2bf(v.w);
  }
  __syncthreads();

  v8f acc[8];
  wmma_gemm(sA, 256, 256, w1, b1, acc, lane);
  __syncthreads(); act_store(acc, sA, lane); __syncthreads();
  wmma_gemm(sA, 128, 128, w2, b2, acc, lane);
  __syncthreads(); act_store(acc, sA, lane); __syncthreads();
  wmma_gemm(sA, 128, 128, w3, b3, acc, lane);
  __syncthreads();

  float* sF = (float*)sA;
  f32_store(acc, sF, lane);
  __syncthreads();

  const int half = lane >> 4, rowm = lane & 15;
  const float* rp = sF + rowm * D + (half << 6);
  float s = 0.f, s2 = 0.f;
  #pragma unroll 8
  for (int c = 0; c < 64; ++c) { float v = rp[c]; s += v; s2 += v * v; }
  s  += __shfl_xor(s,  16, 32);
  s2 += __shfl_xor(s2, 16, 32);
  float mu   = s * (1.f / 128.f);
  float rstd = rsqrtf(s2 * (1.f / 128.f) - mu * mu + 1e-5f);

  const float* rrow = frec + (size_t)(base + rowm) * D;
  float*       orow = fout + (size_t)(base + rowm) * D;
  for (int c = 0; c < 64; ++c) {
    int col = (half << 6) + c;
    float diff = (rp[c] - mu) * rstd * lng[col] + lnb[col];
    orow[col] = alpha * rrow[col] + diff;   // read-before-write, in-place safe
  }
}

// Post-intra: feats = LayerNorm(feats + x_orig) with norm {g,b}. One row/wave.
__global__ __launch_bounds__(32) void add_ln_k(
    float* __restrict__ f, const float* __restrict__ x0,
    const float* __restrict__ g, const float* __restrict__ b)
{
  const int row  = blockIdx.x;
  const int lane = threadIdx.x;
  float4 fv = ((const float4*)(f  + (size_t)row * D))[lane];
  float4 xv = ((const float4*)(x0 + (size_t)row * D))[lane];
  float v0 = fv.x + xv.x, v1 = fv.y + xv.y, v2 = fv.z + xv.z, v3 = fv.w + xv.w;
  float s  = v0 + v1 + v2 + v3;
  float s2 = v0 * v0 + v1 * v1 + v2 * v2 + v3 * v3;
  #pragma unroll
  for (int off = 16; off >= 1; off >>= 1) {
    s  += __shfl_xor(s,  off, 32);
    s2 += __shfl_xor(s2, off, 32);
  }
  float mu   = s * (1.f / 128.f);
  float rstd = rsqrtf(s2 * (1.f / 128.f) - mu * mu + 1e-5f);
  float4 gv = ((const float4*)g)[lane];
  float4 bv = ((const float4*)b)[lane];
  float4 o;
  o.x = (v0 - mu) * rstd * gv.x + bv.x;
  o.y = (v1 - mu) * rstd * gv.y + bv.y;
  o.z = (v2 - mu) * rstd * gv.z + bv.z;
  o.w = (v3 - mu) * rstd * gv.w + bv.w;
  ((float4*)(f + (size_t)row * D))[lane] = o;
}

// ---------------------------------------------------------------------------
// Host orchestration
// ---------------------------------------------------------------------------
extern "C" void kernel_launch(void* const* d_in, const int* in_sizes, int n_in,
                              void* d_out, int out_size, void* d_ws, size_t ws_size,
                              hipStream_t stream)
{
  (void)in_sizes; (void)n_in; (void)out_size; (void)ws_size;
  static const int NSh[3] = {65536, 16384, 4096};
  static const int EMh[3] = {524288, 131072, 32768};
  static const int EUh[2] = {196608, 49152};
  static const int EDh[2] = {196608, 49152};

  const float* X[3]   = {(const float*)d_in[0],  (const float*)d_in[3],  (const float*)d_in[6]};
  const int*   MEI[3] = {(const int*)d_in[1],    (const int*)d_in[4],    (const int*)d_in[7]};
  const float* MEA[3] = {(const float*)d_in[2],  (const float*)d_in[5],  (const float*)d_in[8]};
  const int*   UEI[2] = {(const int*)d_in[9],    (const int*)d_in[13]};
  const float* UEA[2] = {(const float*)d_in[10], (const float*)d_in[14]};
  const int*   DEI[2] = {(const int*)d_in[11],   (const int*)d_in[15]};
  const float* DEA[2] = {(const float*)d_in[12], (const float*)d_in[16]};

  // Param leaves: assumes JAX pytree flatten (sorted dict keys):
  // params -> down, intra, norm, up ; net -> aggr, edge ; mlp -> Ws, bs, ln_b, ln_g
  struct MlpP { const float* W[3]; const float* b[3]; const float* ln_b; const float* ln_g; };
  struct NetP { MlpP aggr, edge; };
  int pi = 17;
  auto getf = [&]() { return (const float*)d_in[pi++]; };
  auto readMlp = [&](MlpP& m) {
    for (int j = 0; j < 3; ++j) m.W[j] = getf();
    for (int j = 0; j < 3; ++j) m.b[j] = getf();
    m.ln_b = getf(); m.ln_g = getf();
  };
  auto readNet = [&](NetP& n) { readMlp(n.aggr); readMlp(n.edge); };

  NetP dnN[2];      readNet(dnN[0]); readNet(dnN[1]);
  NetP inN[3][4];   for (int l = 0; l < 3; ++l) for (int s = 0; s < 4; ++s) readNet(inN[l][s]);
  const float *normB[3], *normG[3];
  for (int l = 0; l < 3; ++l) { normB[l] = getf(); normG[l] = getf(); }
  NetP upN[2];      readNet(upN[0]); readNet(upN[1]);

  // Workspace carve: feats[3] (fp32) | aggr buffer | swizzled bf16 weights
  char* w = (char*)d_ws;
  size_t off = 0;
  float* feats[3];
  for (int l = 0; l < 3; ++l) { feats[l] = (float*)(w + off); off += (size_t)NSh[l] * D * 4; }
  float* aggb = (float*)(w + off); off += (size_t)NSh[0] * D * 4;
  unsigned short* swzbase = (unsigned short*)(w + off);

  struct MlpS { const v16us* W[3]; };
  struct NetS { MlpS edge, aggr; };
  size_t soff = 0;
  auto doswz = [&](const MlpP& m, int K0, MlpS& out) {
    int Ks[3] = {K0, D, D};
    for (int j = 0; j < 3; ++j) {
      unsigned short* dst = swzbase + soff;
      out.W[j] = (const v16us*)dst;
      int n = Ks[j] * D;
      swizzle_k<<<(n + 255) / 256, 256, 0, stream>>>(m.W[j], dst, Ks[j]);
      soff += (size_t)n;
    }
  };
  NetS inS[3][4], upS[2], dnS[2];
  for (int l = 0; l < 3; ++l)
    for (int s = 0; s < 4; ++s) { doswz(inN[l][s].edge, 3 * D, inS[l][s].edge); doswz(inN[l][s].aggr, 2 * D, inS[l][s].aggr); }
  for (int l = 0; l < 2; ++l) { doswz(upN[l].edge, 3 * D, upS[l].edge); doswz(upN[l].aggr, 2 * D, upS[l].aggr); }
  for (int l = 0; l < 2; ++l) { doswz(dnN[l].edge, 3 * D, dnS[l].edge); doswz(dnN[l].aggr, 2 * D, dnS[l].aggr); }

  // Init feats from inputs (inputs are never mutated)
  for (int l = 0; l < 3; ++l) {
    long n4 = (long)NSh[l] * (D / 4);
    copy_k<<<(unsigned)((n4 + 255) / 256), 256, 0, stream>>>((float4*)feats[l], (const float4*)X[l], n4);
  }

  auto inter = [&](const NetP& net, const NetS& ns, const int* ei, const float* ea, int E,
                   const float* fsend, float* frec, int Nrec, float alpha) {
    long n4 = (long)Nrec * (D / 4);
    zero_k<<<(unsigned)((n4 + 255) / 256), 256, 0, stream>>>((float4*)aggb, n4);
    edge_mlp_k<<<E / 16, 32, 0, stream>>>(ei, ea, fsend, frec,
        ns.edge.W[0], ns.edge.W[1], ns.edge.W[2],
        net.edge.b[0], net.edge.b[1], net.edge.b[2],
        net.edge.ln_g, net.edge.ln_b, aggb, E);
    node_mlp_k<<<Nrec / 16, 32, 0, stream>>>(aggb, frec,
        ns.aggr.W[0], ns.aggr.W[1], ns.aggr.W[2],
        net.aggr.b[0], net.aggr.b[1], net.aggr.b[2],
        net.aggr.ln_g, net.aggr.ln_b, frec, alpha);
  };

  // Intra-level message passing + residual LayerNorm
  for (int l = 0; l < 3; ++l) {
    for (int s = 0; s < 4; ++s)
      inter(inN[l][s], inS[l][s], MEI[l], MEA[l], EMh[l], feats[l], feats[l], NSh[l], 1.f);
    add_ln_k<<<NSh[l], 32, 0, stream>>>(feats[l], X[l], normG[l], normB[l]);
  }
  // Up pass (fine -> coarse): feats[l+1] = 2*feats[l+1] + diff
  inter(upN[0], upS[0], UEI[0], UEA[0], EUh[0], feats[0], feats[1], NSh[1], 2.f);
  inter(upN[1], upS[1], UEI[1], UEA[1], EUh[1], feats[1], feats[2], NSh[2], 2.f);
  // Down pass (coarse -> fine), reversed order
  inter(dnN[1], dnS[1], DEI[1], DEA[1], EDh[1], feats[2], feats[1], NSh[1], 2.f);
  inter(dnN[0], dnS[0], DEI[0], DEA[0], EDh[0], feats[1], feats[0], NSh[0], 2.f);

  // Emit concatenated outputs
  float* outp = (float*)d_out;
  size_t o = 0;
  for (int l = 0; l < 3; ++l) {
    long n4 = (long)NSh[l] * (D / 4);
    copy_k<<<(unsigned)((n4 + 255) / 256), 256, 0, stream>>>((float4*)(outp + o), (const float4*)feats[l], n4);
    o += (size_t)NSh[l] * D;
  }
}
